// PairingLoss_34668976013936
// MI455X (gfx1250) — matrix operand order repaired and checked by reading, not compiled, not run on hardware
//
#include <hip/hip_runtime.h>

// ---------------------------------------------------------------------------
// PairingLoss on MI455X (gfx1250, wave32, WMMA)
//   B=64, T=128, H=128, BT=8192
//   Dominant work: neg[i] = sum_j exp(dot(x_i, x_j))  (8192x8192x128 GEMM)
//   done with split-f16 WMMA (hi*hi + hi*lo + lo*hi, f32 accumulate).
// ---------------------------------------------------------------------------

#define BDIM 64
#define TDIM 128
#define HDIM 128
#define BT   8192          // BDIM * TDIM
#define NEG_FILL (-10000.0f)

#define MTILE  128         // rows per block   (8 waves x 16 rows)
#define NCHUNK 1024        // columns per block (grid.y chunk)
#define NPART  (BT / NCHUNK)  // 8 column chunks -> 8 partial neg arrays

typedef _Float16 v16h __attribute__((ext_vector_type(16)));
typedef _Float16 v8h  __attribute__((ext_vector_type(8)));
typedef float    v8f  __attribute__((ext_vector_type(8)));

// ---------------------------------------------------------------------------
// Prep: split f32 -> (hi f16, lo f16); adj[i] = dot(row_i, row_{i+1}) in-batch
// One wave (32 lanes) per token row. grid = BT/8 blocks of 256 threads.
// ---------------------------------------------------------------------------
__global__ __launch_bounds__(256)
void pl_prep(const float* __restrict__ x,
             _Float16* __restrict__ xhi,
             _Float16* __restrict__ xlo,
             float* __restrict__ adj) {
  const int wave = threadIdx.x >> 5;
  const int lane = threadIdx.x & 31;
  const int i = blockIdx.x * 8 + wave;           // global token row
  const int t = i & (TDIM - 1);
  const bool hasNext = (t < TDIM - 1);

  const float* row = x + (size_t)i * HDIM;
  const float* nxt = row + HDIM;

  float sa = 0.f;
#pragma unroll
  for (int c = 0; c < HDIM / 32; ++c) {
    const int h = c * 32 + lane;
    const float v = row[h];
    const _Float16 hi = (_Float16)v;
    const _Float16 lo = (_Float16)(v - (float)hi);
    xhi[(size_t)i * HDIM + h] = hi;
    xlo[(size_t)i * HDIM + h] = lo;
    if (hasNext) sa += v * nxt[h];
  }
#pragma unroll
  for (int m = 16; m >= 1; m >>= 1) sa += __shfl_xor(sa, m, 32);
  if (lane == 0) adj[i] = hasNext ? sa : 0.f;
}

// ---------------------------------------------------------------------------
// Fused GEMM + exp + row-sum.
//   block = 256 threads = 8 waves; wave w owns rows rowBase..rowBase+15.
//   grid  = (BT/MTILE, NPART).  Each block sweeps NCHUNK columns and writes
//   negpart[blockIdx.y][row] exactly once (deterministic, no atomics).
// A-fragment layout (f16 16x32, ISA 7.12.2): lane<16 row=lane K={0..7,16..23},
//   lane>=16 row=lane-16 K={8..15,24..31}.
// B-fragment layout (f16 32x16): lane<16 col=lane K={0..7,16..23}, lane>=16 K={8..15,24..31}.
// ---------------------------------------------------------------------------
__global__ __launch_bounds__(256)
void pl_gemm_exp(const _Float16* __restrict__ xhi,
                 const _Float16* __restrict__ xlo,
                 float* __restrict__ negpart) {
  const int wave = threadIdx.x >> 5;
  const int lane = threadIdx.x & 31;
  const int lrow = lane & 15;
  const int lsel = lane >> 4;
  const int rowBase = blockIdx.x * MTILE + wave * 16;
  const int colBase = blockIdx.y * NCHUNK;

  // Load A fragments for this wave's 16 rows: 4 K-chunks, hi and lo.
  v16h ahi[4], alo[4];
  {
    const _Float16* ph = xhi + (size_t)(rowBase + lrow) * HDIM;
    const _Float16* pl = xlo + (size_t)(rowBase + lrow) * HDIM;
#pragma unroll
    for (int kc = 0; kc < 4; ++kc) {
      const int base = kc * 32 + lsel * 8;   // K blocks {base..base+7, base+16..base+23}
      union { v16h v; v8h h[2]; } th, tl;
      th.h[0] = *(const v8h*)(ph + base);
      th.h[1] = *(const v8h*)(ph + base + 16);
      tl.h[0] = *(const v8h*)(pl + base);
      tl.h[1] = *(const v8h*)(pl + base + 16);
      ahi[kc] = th.v;
      alo[kc] = tl.v;
    }
  }

  float rowsum[8];
#pragma unroll
  for (int r = 0; r < 8; ++r) rowsum[r] = 0.f;

  for (int nt = 0; nt < NCHUNK / 16; ++nt) {
    const int col = colBase + nt * 16 + lrow;
    const _Float16* ph = xhi + (size_t)col * HDIM;
    const _Float16* pl = xlo + (size_t)col * HDIM;
    v8f c = {};
#pragma unroll
    for (int kc = 0; kc < 4; ++kc) {
      const int base = kc * 32 + lsel * 8;   // K blocks {base..base+7, base+16..base+23}
      union { v16h v; v8h h[2]; } bh, bl;
      bh.h[0] = *(const v8h*)(ph + base);
      bh.h[1] = *(const v8h*)(ph + base + 16);
      bl.h[0] = *(const v8h*)(pl + base);
      bl.h[1] = *(const v8h*)(pl + base + 16);
      // split-precision product: hi*hi + hi*lo + lo*hi  (f32 accumulate)
      c = __builtin_amdgcn_wmma_f32_16x16x32_f16(false, ahi[kc], false, bh.v,
                                                 (short)0, c, false, false);
      c = __builtin_amdgcn_wmma_f32_16x16x32_f16(false, ahi[kc], false, bl.v,
                                                 (short)0, c, false, false);
      c = __builtin_amdgcn_wmma_f32_16x16x32_f16(false, alo[kc], false, bh.v,
                                                 (short)0, c, false, false);
    }
    // C/D layout: c[r] in lane L -> row r + 8*(L>=16), col colBase+nt*16+(L&15)
#pragma unroll
    for (int r = 0; r < 8; ++r) rowsum[r] += __expf(c[r]);
  }

  // Reduce each row across its 16-lane group.
#pragma unroll
  for (int r = 0; r < 8; ++r) {
    float v = rowsum[r];
    v += __shfl_xor(v, 1, 32);
    v += __shfl_xor(v, 2, 32);
    v += __shfl_xor(v, 4, 32);
    v += __shfl_xor(v, 8, 32);
    rowsum[r] = v;
  }
  if (lrow == 0) {
    // lane 0 -> rows rowBase+0..7 ; lane 16 -> rows rowBase+8..15
    float* dst = negpart + (size_t)blockIdx.y * BT + rowBase + lsel * 8;
#pragma unroll
    for (int r = 0; r < 8; ++r) dst[r] = rowsum[r];
  }
}

// ---------------------------------------------------------------------------
// Finalize: loss = mean over valid tokens of log(neg_i) - pos_sum_i.
// (Diagonal NEG_FILL fix only affects masked tokens, which are excluded.)
// Single block of 1024 threads, deterministic tree reduction.
// ---------------------------------------------------------------------------
__global__ __launch_bounds__(1024)
void pl_finalize(const int* __restrict__ dia,
                 const float* __restrict__ negpart,
                 const float* __restrict__ adj,
                 float* __restrict__ out) {
  __shared__ float s_sum[1024];
  __shared__ float s_cnt[1024];
  float ls = 0.f, lc = 0.f;
  for (int i = threadIdx.x; i < BT; i += 1024) {
    const int b = i >> 7;            // i / TDIM
    const int t = i & (TDIM - 1);
    const int dl = dia[b];
    const bool m2 = (t >= dl - 1);   // pad or last real token
    if (!m2) {
      float neg = 0.f;
#pragma unroll
      for (int c = 0; c < NPART; ++c) neg += negpart[(size_t)c * BT + i];
      float pos = 0.f;
      if (t > 0) pos += adj[i - 1];
      if (t < TDIM - 1) pos += adj[i];
      ls += __logf(neg) - pos;
      lc += 1.f;
    }
  }
  s_sum[threadIdx.x] = ls;
  s_cnt[threadIdx.x] = lc;
  __syncthreads();
  for (int s = 512; s > 0; s >>= 1) {
    if ((int)threadIdx.x < s) {
      s_sum[threadIdx.x] += s_sum[threadIdx.x + s];
      s_cnt[threadIdx.x] += s_cnt[threadIdx.x + s];
    }
    __syncthreads();
  }
  if (threadIdx.x == 0) out[0] = s_sum[0] / s_cnt[0];
}

// ---------------------------------------------------------------------------
extern "C" void kernel_launch(void* const* d_in, const int* in_sizes, int n_in,
                              void* d_out, int out_size, void* d_ws, size_t ws_size,
                              hipStream_t stream) {
  const float* x  = (const float*)d_in[0];   // hidden_states [64,128,128] f32
  // d_in[1] = mask (bool) -- unused; reconstructed from dia_lens instead
  const int* dia  = (const int*)d_in[2];     // dia_lens [64]
  float* out = (float*)d_out;

  // Workspace carve-up (~4.3 MB total).
  char* ws = (char*)d_ws;
  _Float16* xhi    = (_Float16*)(ws);                       // 2 MB
  _Float16* xlo    = (_Float16*)(ws + (size_t)2 * 1024 * 1024);   // 2 MB
  float*    adj    = (float*)(ws + (size_t)4 * 1024 * 1024);      // 32 KB
  float*    negpart= (float*)(ws + (size_t)4 * 1024 * 1024 + 32 * 1024); // 256 KB

  pl_prep<<<BT / 8, 256, 0, stream>>>(x, xhi, xlo, adj);
  pl_gemm_exp<<<dim3(BT / MTILE, NPART), 256, 0, stream>>>(xhi, xlo, negpart);
  pl_finalize<<<1, 1024, 0, stream>>>(dia, negpart, adj, out);
}
